// GNNClassifier_88648124990101
// MI455X (gfx1250) — compile-verified, matrix operand order checked by reading, not run on hardware
//
#include <hip/hip_runtime.h>

// ---------------------------------------------------------------------------
// GNN (GraphSAGE x2 + mean-pool + linear head) for MI455X (gfx1250, wave32).
//
// Memory-bound parts (edge gather + segment_sum) use coalesced per-
// (edge,feature) global_atomic_add_f32.  The node-feature projections
// (mean @ Wl^T + x @ Wr^T) use v_wmma_f32_16x16x32_f16: one wave owns a
// 16-node row tile, loops over the 4 column tiles of HID=64, and chains
// WMMA accumulations over K.  N_NODES % 16 == 0, so there is no row tail:
// no clamps, no store guards, EXEC stays all-1s through the whole kernel.
// ---------------------------------------------------------------------------

#define N_NODES  100000
#define N_EDGES  3200000
#define N_GRAPHS 4096
#define EMB      32
#define HID      64
#define NCLS     10

typedef __attribute__((ext_vector_type(16))) _Float16 v16h;
typedef __attribute__((ext_vector_type(8)))  float    v8f;

// Build a 16-element f16 A/B fragment from four float4 chunks (scaled).
// Element e maps to k = half*8 + e + (e>=8 ? 8 : 0); chunks are the two
// 8-float runs [half*8, half*8+8) and [16+half*8, 16+half*8+8).
__device__ __forceinline__ v16h pack_frag(float4 c0, float4 c1,
                                          float4 c2, float4 c3, float s)
{
    v16h r;
    r[0]  = (_Float16)(c0.x * s); r[1]  = (_Float16)(c0.y * s);
    r[2]  = (_Float16)(c0.z * s); r[3]  = (_Float16)(c0.w * s);
    r[4]  = (_Float16)(c1.x * s); r[5]  = (_Float16)(c1.y * s);
    r[6]  = (_Float16)(c1.z * s); r[7]  = (_Float16)(c1.w * s);
    r[8]  = (_Float16)(c2.x * s); r[9]  = (_Float16)(c2.y * s);
    r[10] = (_Float16)(c2.z * s); r[11] = (_Float16)(c2.w * s);
    r[12] = (_Float16)(c3.x * s); r[13] = (_Float16)(c3.y * s);
    r[14] = (_Float16)(c3.z * s); r[15] = (_Float16)(c3.w * s);
    return r;
}

// Load one 16-wide K fragment (for row `base` of a [*,K] f32 matrix) as f16.
// `qbase` = float4 pointer to the row, `ks` = 32-wide K step, `half` = lane>>4.
__device__ __forceinline__ v16h load_frag(const float4* __restrict__ q,
                                          int ks, int half, float s)
{
    const int o = ks * 8 + half * 2;
    return pack_frag(q[o], q[o + 1], q[o + 4], q[o + 5], s);
}

// ---------------------------------------------------------------------------
// x0[node, f] = embed_w[x_idx[node], f]
__global__ void embed_gather(const int* __restrict__ x_idx,
                             const float* __restrict__ embed_w,
                             float* __restrict__ x0)
{
    int tid = blockIdx.x * blockDim.x + threadIdx.x;
    if (tid >= N_NODES * EMB) return;
    int node = tid >> 5;          // /EMB (EMB==32)
    int f    = tid & (EMB - 1);
    x0[tid] = embed_w[x_idx[node] * EMB + f];
}

// cnt[dst[e]] += 1  (in-degree; reused by both SAGE layers)
__global__ void edge_count(const int* __restrict__ dst, float* __restrict__ cnt)
{
    int e = blockIdx.x * blockDim.x + threadIdx.x;
    if (e >= N_EDGES) return;
    atomicAdd(&cnt[dst[e]], 1.0f);
}

// agg[dst[e], f] += x[src[e], f]   -- thread per (edge, feature), feature-major
// so consecutive lanes hit consecutive features: coalesced loads and
// coalesced atomic traffic.
template <int F>
__global__ void edge_scatter(const int* __restrict__ src,
                             const int* __restrict__ dst,
                             const float* __restrict__ x,
                             float* __restrict__ agg)
{
    int tid = blockIdx.x * blockDim.x + threadIdx.x;
    if (tid >= N_EDGES * F) return;
    int e = tid / F;
    int f = tid & (F - 1);
    int s = src[e];
    int d = dst[e];
    atomicAdd(&agg[d * F + f], x[s * F + f]);
}

// ---------------------------------------------------------------------------
// out[i, :] = relu( (agg[i,:]/max(cnt,1)) @ Wl^T + bl + x[i,:] @ Wr^T )
//
// WMMA mapping (wave32, v_wmma_f32_16x16x32_f16):
//   A 16x32 f16: lane = {half = lane>>4, m = lane&15}, element e holds
//       K = half*8 + e + (e>=8 ? 8 : 0)     (ISA 7.12.2 16-bit A layout)
//   B 32x16 f16: lane column n = lane&15, same K mapping; B[k][n] = W[n][k].
//   C/D 16x16 f32: VGPR d -> row = d + half*8, col = m.
template <int K>   // K = EMB(32) or HID(64); output width fixed at HID=64
__global__ void sage_linear_wmma(const float* __restrict__ agg,
                                 const float* __restrict__ cnt,
                                 const float* __restrict__ xin,
                                 const float* __restrict__ Wl,   // [HID, K]
                                 const float* __restrict__ bl,   // [HID]
                                 const float* __restrict__ Wr,   // [HID, K]
                                 float* __restrict__ out)        // [N, HID]
{
    const int lane  = threadIdx.x & 31;
    const int wave  = threadIdx.x >> 5;
    const int wpb   = blockDim.x >> 5;
    const int tile  = blockIdx.x * wpb + wave;       // 16-node row tile
    const int row0  = tile * 16;
    if (row0 >= N_NODES) return;                     // wave-uniform exit

    const int half = lane >> 4;
    const int m    = lane & 15;
    const int rowA = row0 + m;                       // always in range (N%16==0)

    const float inv = 1.0f / fmaxf(cnt[rowA], 1.0f);

    constexpr int KS = K / 32;
    const float4* aggq = (const float4*)(agg + rowA * K);
    const float4* xq   = (const float4*)(xin + rowA * K);

    v16h aM[KS], aX[KS];
    #pragma unroll
    for (int ks = 0; ks < KS; ++ks) {
        aM[ks] = load_frag(aggq, ks, half, inv);
        aX[ks] = load_frag(xq,   ks, half, 1.0f);
    }

    #pragma unroll
    for (int nt = 0; nt < HID / 16; ++nt) {
        const int ncol = nt * 16 + m;                // this lane's B column
        const float4* wlq = (const float4*)(Wl + ncol * K);
        const float4* wrq = (const float4*)(Wr + ncol * K);

        v8f acc = {};
        #pragma unroll
        for (int ks = 0; ks < KS; ++ks) {
            v16h bL = load_frag(wlq, ks, half, 1.0f);
            v16h bR = load_frag(wrq, ks, half, 1.0f);
            acc = __builtin_amdgcn_wmma_f32_16x16x32_f16(
                false, aM[ks], false, bL, (short)0, acc, false, false);
            acc = __builtin_amdgcn_wmma_f32_16x16x32_f16(
                false, aX[ks], false, bR, (short)0, acc, false, false);
        }

        const float bias = bl[ncol];
        float* op = out + (row0 + half * 8) * HID + ncol;
        #pragma unroll
        for (int d = 0; d < 8; ++d)
            op[d * HID] = fmaxf(acc[d] + bias, 0.0f);   // no guard: no tail
    }
}

// ---------------------------------------------------------------------------
// pooled[batch[i], f] += h2[i, f]
__global__ void pool_features(const int* __restrict__ batch,
                              const float* __restrict__ h2,
                              float* __restrict__ pooled)
{
    int tid = blockIdx.x * blockDim.x + threadIdx.x;
    if (tid >= N_NODES * HID) return;
    int node = tid >> 6;           // /HID (HID==64)
    int f    = tid & (HID - 1);
    atomicAdd(&pooled[batch[node] * HID + f], h2[tid]);
}

__global__ void pool_count(const int* __restrict__ batch, float* __restrict__ gcnt)
{
    int i = blockIdx.x * blockDim.x + threadIdx.x;
    if (i >= N_NODES) return;
    atomicAdd(&gcnt[batch[i]], 1.0f);
}

// out[g, c] = bout[c] + (pooled[g,:]/max(gcnt,1)) . Wout[c,:]
__global__ void final_head(const float* __restrict__ pooled,
                           const float* __restrict__ gcnt,
                           const float* __restrict__ Wout,   // [NCLS, HID]
                           const float* __restrict__ bout,
                           float* __restrict__ out)
{
    int tid = blockIdx.x * blockDim.x + threadIdx.x;
    if (tid >= N_GRAPHS * NCLS) return;
    int g = tid / NCLS;
    int c = tid % NCLS;
    float inv = 1.0f / fmaxf(gcnt[g], 1.0f);
    float acc = bout[c];
    #pragma unroll
    for (int h = 0; h < HID; ++h)
        acc += pooled[g * HID + h] * inv * Wout[c * HID + h];
    out[tid] = acc;
}

// ---------------------------------------------------------------------------
extern "C" void kernel_launch(void* const* d_in, const int* in_sizes, int n_in,
                              void* d_out, int out_size, void* d_ws, size_t ws_size,
                              hipStream_t stream)
{
    const int*   x_idx   = (const int*)  d_in[0];            // [N_NODES]
    const int*   ei      = (const int*)  d_in[1];            // [2, N_EDGES]
    const int*   batch   = (const int*)  d_in[2];            // [N_NODES]
    const float* embed_w = (const float*)d_in[3];            // [64, EMB]
    const float* W1l     = (const float*)d_in[4];
    const float* b1      = (const float*)d_in[5];
    const float* W1r     = (const float*)d_in[6];
    const float* W2l     = (const float*)d_in[7];
    const float* b2      = (const float*)d_in[8];
    const float* W2r     = (const float*)d_in[9];
    const float* Wout    = (const float*)d_in[10];
    const float* bout    = (const float*)d_in[11];
    float*       out     = (float*)d_out;

    const int* src = ei;
    const int* dst = ei + N_EDGES;

    // Workspace layout (floats). agg2 aliases the x0+agg1 region (both dead
    // after layer-1 linear) -> total ~78 MB.
    float* cnt    = (float*)d_ws;                 // N_NODES
    float* x0     = cnt    + N_NODES;             // N_NODES*EMB
    float* agg1   = x0     + N_NODES * EMB;       // N_NODES*EMB
    float* agg2   = x0;                           // N_NODES*HID (aliases x0+agg1)
    float* h1     = agg1   + N_NODES * EMB;       // N_NODES*HID
    float* h2     = h1     + N_NODES * HID;       // N_NODES*HID
    float* pooled = h2     + N_NODES * HID;       // N_GRAPHS*HID
    float* gcnt   = pooled + N_GRAPHS * HID;      // N_GRAPHS

    const int B = 256;

    // Zero accumulation buffers (workspace is poisoned, not re-zeroed).
    hipMemsetAsync(cnt,    0, sizeof(float) * N_NODES,        stream);
    hipMemsetAsync(agg1,   0, sizeof(float) * N_NODES * EMB,  stream);
    hipMemsetAsync(pooled, 0, sizeof(float) * N_GRAPHS * HID, stream);
    hipMemsetAsync(gcnt,   0, sizeof(float) * N_GRAPHS,       stream);

    // Embedding gather.
    embed_gather<<<(N_NODES * EMB + B - 1) / B, B, 0, stream>>>(x_idx, embed_w, x0);

    // In-degree (shared by both layers).
    edge_count<<<(N_EDGES + B - 1) / B, B, 0, stream>>>(dst, cnt);

    // ---- Layer 1 ----
    edge_scatter<EMB><<<(N_EDGES * EMB + B - 1) / B, B, 0, stream>>>(src, dst, x0, agg1);
    {
        const int tiles  = N_NODES / 16;          // 6250, exact
        const int blocks = (tiles + 7) / 8;       // 8 waves/block
        sage_linear_wmma<EMB><<<blocks, 256, 0, stream>>>(agg1, cnt, x0, W1l, b1, W1r, h1);
    }

    // ---- Layer 2 ----  (agg2 region re-zeroed after it stops aliasing live data)
    hipMemsetAsync(agg2, 0, sizeof(float) * N_NODES * HID, stream);
    edge_scatter<HID><<<(int)(((long long)N_EDGES * HID + B - 1) / B), B, 0, stream>>>(src, dst, h1, agg2);
    {
        const int tiles  = N_NODES / 16;
        const int blocks = (tiles + 7) / 8;
        sage_linear_wmma<HID><<<blocks, 256, 0, stream>>>(agg2, cnt, h1, W2l, b2, W2r, h2);
    }

    // ---- Mean pool + head ----
    pool_features<<<(N_NODES * HID + B - 1) / B, B, 0, stream>>>(batch, h2, pooled);
    pool_count<<<(N_NODES + B - 1) / B, B, 0, stream>>>(batch, gcnt);
    final_head<<<(N_GRAPHS * NCLS + B - 1) / B, B, 0, stream>>>(pooled, gcnt, Wout, bout, out);
}
